// FinalPassLSTMWithLastK_19378892439730
// MI455X (gfx1250) — matrix operand order repaired and checked by reading, not compiled, not run on hardware
//
#include <hip/hip_runtime.h>
#include <hip/hip_bf16.h>

typedef __attribute__((ext_vector_type(16))) _Float16 v16h;
typedef __attribute__((ext_vector_type(8)))  float    v8f;

constexpr int Bn   = 256;
constexpr int Tn   = 2048;
constexpr int Fn   = 12;
constexpr int EEDn = 6;
constexpr int REDn = 3;
constexpr int Dn   = Fn + EEDn + REDn;   // 21
constexpr int Hn   = 96;
constexpr int Kn   = 3;
constexpr int G4H  = 4 * Hn;             // 384
constexpr int KPAD = 128;                // padded K for [x|h] = 117 -> 128
constexpr int ROWS = 16;                 // batch rows per workgroup
constexpr int NWAVES = G4H / 16;         // 24 gate-column tiles, 1 per wave
constexpr int THREADS = NWAVES * 32;     // 768

// Branch-free activations on native v_exp_f32 / v_rcp_f32.
__device__ __forceinline__ float fsigmoid(float x) {
    return __builtin_amdgcn_rcpf(1.f + __builtin_amdgcn_exp2f(-1.4426950408889634f * x));
}
__device__ __forceinline__ float ftanh(float x) {
    return 1.f - 2.f * __builtin_amdgcn_rcpf(1.f + __builtin_amdgcn_exp2f(2.8853900817779268f * x));
}

// Workgroup barrier that orders ONLY LDS traffic (DScnt), leaving in-flight
// global loads (the software-pipelined x/index fetches) outstanding across the
// barrier. __syncthreads() would insert s_wait_loadcnt_dscnt 0x0 and drain them.
__device__ __forceinline__ void wg_barrier_lds() {
    asm volatile(
        "s_wait_dscnt 0x0\n\t"
        "s_barrier_signal -1\n\t"
        "s_barrier_wait -1"
        ::: "memory");
}

// Persistent LSTM kernel: one workgroup owns 16 batch rows for all T steps.
__global__ __launch_bounds__(THREADS) void lstm_persistent(
    const float* __restrict__ seq, const int* __restrict__ ev,
    const int* __restrict__ rs, const int* __restrict__ lengths,
    const float* __restrict__ event_emb, const float* __restrict__ result_emb,
    const float* __restrict__ W_ih, const float* __restrict__ W_hh,
    const float* __restrict__ b_ih, const float* __restrict__ b_hh,
    float* __restrict__ lastk /* [B][K][H] */)
{
    __shared__ _Float16 xh[ROWS][KPAD];     // [ x_t (21) | h (96) | zero pad (11) ]
    __shared__ float    gates[ROWS][G4H];   // raw gate pre-activations

    const int tid  = threadIdx.x;
    const int lane = tid & 31;
    const int wave = tid >> 5;
    const int row0 = blockIdx.x * ROWS;

    for (int i = tid; i < ROWS * KPAD; i += THREADS)
        ((_Float16*)xh)[i] = (_Float16)0.f;
    for (int i = tid; i < ROWS * Kn * Hn; i += THREADS)
        lastk[(size_t)row0 * Kn * Hn + i] = 0.f;

    // ---- one-time preload: per-wave B fragments of combined weight ----
    // Wc[k][n] = k<21 ? W_ih[n][k] : (k<117 ? W_hh[n][k-21] : 0)
    const int n16  = wave * 16;
    const int ncol = n16 + (lane & 15);
    const int koff = (lane & 16) ? 8 : 0;
    v16h bfrag[4];
    #pragma unroll
    for (int kc = 0; kc < 4; ++kc) {
        #pragma unroll
        for (int e = 0; e < 16; ++e) {
            int k = kc * 32 + koff + e + (e >= 8 ? 8 : 0);
            float w;
            if (k < Dn)           w = W_ih[ncol * Dn + k];
            else if (k < Dn + Hn) w = W_hh[ncol * Hn + (k - Dn)];
            else                  w = 0.f;
            bfrag[kc][e] = (_Float16)w;
        }
    }
    const float bias = b_ih[ncol] + b_hh[ncol];

    // ---- per-thread persistent cell state: 2 cells/thread ----
    float creg[2];
    int cm[2], cd[2], clen[2];
    #pragma unroll
    for (int i = 0; i < 2; ++i) {
        int idx = tid + i * THREADS;
        cm[i] = idx / Hn; cd[i] = idx % Hn;
        clen[i] = lengths[row0 + cm[i]];
        creg[i] = 0.f;
    }

    const int arow  = lane & 15;
    const int akoff = (lane & 16) ? 8 : 0;
    const int m2off = (lane & 16) ? 8 : 0;

    // ---- depth-2 software-pipelined x staging ----
    //   sk <  12        : seq element       (address is t-indexed, no gather)
    //   12 <= sk < 18   : event_emb gather  (index loaded 1 iteration ahead)
    //   18 <= sk < 21   : result_emb gather (index loaded 1 iteration ahead)
    const bool stage = tid < ROWS * Dn;
    const int  sm = tid / Dn, sk = tid % Dn;
    const size_t btbase = (size_t)(row0 + sm) * Tn;

    const int*   idxp = nullptr;   // nullptr => direct seq path
    const float* src  = seq;
    int mul = 0, off = 0;
    if (sk >= Fn) {
        if (sk < Fn + EEDn) { idxp = ev; src = event_emb; mul = EEDn; off = sk - Fn; }
        else                { idxp = rs; src = result_emb; mul = REDn; off = sk - Fn - EEDn; }
    }

    int   nidx = 0;    // gather index for the NEXT staged step
    float xreg = 0.f;  // x value for the CURRENT staged step
    if (stage) {
        if (idxp) {
            int i0 = idxp[btbase];                 // index for t=0
            xreg = src[i0 * mul + off];            // value for t=0
            nidx = idxp[btbase + 1];               // index for t=1 (Tn >= 2)
        } else {
            xreg = seq[btbase * Fn + sk];          // value for t=0
        }
    }

    for (int t = 0; t < Tn; ++t) {
        // commit pre-loaded x_t (cvt + ds_store only; load issued last iter)
        if (stage) xh[sm][sk] = (_Float16)xreg;
        wg_barrier_lds();  // sync1: x staged, prev h update visible (DScnt only)

        // ---- gates[:, n16:n16+16] = [x|h] @ Wc tile ----
        v16h a0, a1, a2, a3;
        {
            const _Float16* rp = &xh[arow][akoff];
            #pragma unroll
            for (int e = 0; e < 8; ++e) { a0[e] = rp[e];       a0[e + 8] = rp[16 + e]; }
            #pragma unroll
            for (int e = 0; e < 8; ++e) { a1[e] = rp[32 + e];  a1[e + 8] = rp[48 + e]; }
            #pragma unroll
            for (int e = 0; e < 8; ++e) { a2[e] = rp[64 + e];  a2[e + 8] = rp[80 + e]; }
            #pragma unroll
            for (int e = 0; e < 8; ++e) { a3[e] = rp[96 + e];  a3[e + 8] = rp[112 + e]; }
        }
        // ballast: keep all four fragments live -> 8 back-to-back ds_load_b128
        asm volatile("" : "+v"(a0), "+v"(a1), "+v"(a2), "+v"(a3));

        v8f acc;
        #pragma unroll
        for (int r = 0; r < 8; ++r) acc[r] = bias;
        acc = __builtin_amdgcn_wmma_f32_16x16x32_f16(false, a0, false, bfrag[0], (short)0, acc, false, false);
        acc = __builtin_amdgcn_wmma_f32_16x16x32_f16(false, a1, false, bfrag[1], (short)0, acc, false, false);
        acc = __builtin_amdgcn_wmma_f32_16x16x32_f16(false, a2, false, bfrag[2], (short)0, acc, false, false);
        acc = __builtin_amdgcn_wmma_f32_16x16x32_f16(false, a3, false, bfrag[3], (short)0, acc, false, false);

        #pragma unroll
        for (int r = 0; r < 8; ++r)
            gates[m2off + r][n16 + (lane & 15)] = acc[r];

        // issue x_{t+1} value load (index already resident) and t+2 index load;
        // these now stay outstanding ACROSS the barrier (DScnt-only wait) and
        // are consumed a full iteration later -> no loadcnt drain on the path
        if (stage && t + 1 < Tn) {
            size_t bt1 = btbase + (t + 1);
            if (idxp) {
                xreg = src[nidx * mul + off];          // gather for t+1
                if (t + 2 < Tn) nidx = idxp[bt1 + 1];  // index for t+2
            } else {
                xreg = seq[bt1 * Fn + sk];
                if (t + 2 < Tn)
                    __builtin_prefetch(&seq[(bt1 + 1) * Fn + sk], 0, 1);
            }
        }
        wg_barrier_lds();  // sync2: gates visible (DScnt only)

        // ---- activations + state update (gate order i,f,g,o) ----
        #pragma unroll
        for (int i = 0; i < 2; ++i) {
            int m = cm[i], d = cd[i], len = clen[i];
            if (t < len) {  // packed-seq: freeze state past length
                float si = fsigmoid(gates[m][d]);
                float sf = fsigmoid(gates[m][Hn + d]);
                float tg = ftanh(gates[m][2 * Hn + d]);
                float so = fsigmoid(gates[m][3 * Hn + d]);
                float cn = sf * creg[i] + si * tg;
                float hn = so * ftanh(cn);
                creg[i] = cn;
                xh[m][Dn + d] = (_Float16)hn;   // h for next step's matmul
                int j = t - (len - Kn);          // last-K window recording
                if (j >= 0)
                    lastk[((size_t)(row0 + m) * Kn + j) * Hn + d] = hn;
            }
        }
        // next iteration's sync1 orders these LDS writes vs. WMMA reads
    }
    // S_ENDPGM performs an implicit wait-idle; trailing lastk stores complete.
}

// Tiny head: h_lastk = relu(lastk @ W_mlpT + b_mlp); out = [h_fin|h_lastk] @ W_fcT + b_fc
__global__ __launch_bounds__(128) void head_kernel(
    const float* __restrict__ lastk, const float* __restrict__ W_mlp,
    const float* __restrict__ b_mlp, const float* __restrict__ W_fc,
    const float* __restrict__ b_fc, float* __restrict__ out)
{
    __shared__ float lk[Kn * Hn];
    __shared__ float hl[Hn];
    const int b = blockIdx.x, tid = threadIdx.x;
    for (int i = tid; i < Kn * Hn; i += 128)
        lk[i] = lastk[(size_t)b * Kn * Hn + i];
    __syncthreads();
    if (tid < Hn) {
        float s = b_mlp[tid];
        #pragma unroll 4
        for (int p = 0; p < Kn * Hn; ++p) s += W_mlp[tid * Kn * Hn + p] * lk[p];
        hl[tid] = s > 0.f ? s : 0.f;
    }
    __syncthreads();
    if (tid < 2) {
        float s = b_fc[tid];
        const float* w = &W_fc[tid * 2 * Hn];
        for (int p = 0; p < Hn; ++p) s += w[p] * lk[(Kn - 1) * Hn + p]; // h_fin
        for (int p = 0; p < Hn; ++p) s += w[Hn + p] * hl[p];
        out[b * 2 + tid] = s;
    }
}

extern "C" void kernel_launch(void* const* d_in, const int* in_sizes, int n_in,
                              void* d_out, int out_size, void* d_ws, size_t ws_size,
                              hipStream_t stream)
{
    const float* seq        = (const float*)d_in[0];
    const int*   ev         = (const int*)  d_in[1];
    const int*   rs         = (const int*)  d_in[2];
    const int*   lengths    = (const int*)  d_in[3];
    const float* event_emb  = (const float*)d_in[4];
    const float* result_emb = (const float*)d_in[5];
    const float* W_ih       = (const float*)d_in[6];
    const float* W_hh       = (const float*)d_in[7];
    const float* b_ih       = (const float*)d_in[8];
    const float* b_hh       = (const float*)d_in[9];
    const float* W_mlp      = (const float*)d_in[10];
    const float* b_mlp      = (const float*)d_in[11];
    const float* W_fc       = (const float*)d_in[12];
    const float* b_fc       = (const float*)d_in[13];

    float* lastk = (float*)d_ws;  // [B][K][H] f32 = 294,912 bytes

    lstm_persistent<<<Bn / ROWS, THREADS, 0, stream>>>(
        seq, ev, rs, lengths, event_emb, result_emb,
        W_ih, W_hh, b_ih, b_hh, lastk);

    head_kernel<<<Bn, 128, 0, stream>>>(lastk, W_mlp, b_mlp, W_fc, b_fc, (float*)d_out);
}
